// SmoothMinMaxNetworkWithMLP_45294725104125
// MI455X (gfx1250) — matrix operand order repaired and checked by reading, not compiled, not run on hardware
//
#include <hip/hip_runtime.h>
#include <hip/hip_bf16.h>

// Problem constants (from reference): B=1048576, N=64, K=8, HK=16, H=64, NM=32
#define NB  64      // x feature dim
#define NK  8       // k heads
#define NHK 16      // h per head
#define NH  64      // MLP hidden
#define NNM 32      // MLP input (x[:,32:64])

#define LOG2E 1.4426950408889634f

typedef __attribute__((ext_vector_type(16))) __bf16 v16bf;
typedef __attribute__((ext_vector_type(8)))  float  v8f;

static __device__ __forceinline__ v8f wmma_bf16(v16bf a, v16bf b, v8f c) {
  return __builtin_amdgcn_wmma_f32_16x16x32_bf16(
      /*neg_a=*/false, a, /*neg_b=*/false, b,
      /*c_mod=*/(short)0, c, /*reuse_a=*/false, /*reuse_b=*/false);
}

// ---------------------------------------------------------------------------
// Prep: build bf16 B-fragments for the main weight matrix w[k,h,n]
// (w = exp(z) for n<32, z otherwise) and for W1 of the MLP.
//
// B-fragment layout used here (contiguous K, same as A) for v_wmma_f32_16x16x32_bf16 (B is Kdim x 16 cols):
//   lane l: column N = l&15; element i holds Kdim = (l>=16 ? 16 : 0) + i.
// wbuf tile index (j, c): j = k head (N-tile), c = K chunk (n 0..31 / 32..63).
//   wbuf[((j*2+c)*32 + l)*16 + i] = bf16( w[j][l&15][c*32 + (l>=16?16:0) + i] )
// w1buf tile q (h-tile of H=64):
//   w1buf[(q*32 + l)*16 + i]      = bf16( W1[q*16 + (l&15)][(l>=16?16:0) + i] )
// ---------------------------------------------------------------------------
__global__ void __launch_bounds__(256)
prep_kernel(const float* __restrict__ z, const float* __restrict__ W1,
            __bf16* __restrict__ wbuf, __bf16* __restrict__ w1buf) {
  int tid = blockIdx.x * blockDim.x + threadIdx.x;
  if (tid < NK * 2 * 32 * 16) {            // 8192 elements of wbuf
    int i = tid & 15;
    int l = (tid >> 4) & 31;
    int c = (tid >> 9) & 1;
    int j = tid >> 10;
    int h = l & 15;
    int n = c * 32 + ((l & 16) ? 16 : 0) + i;
    float v = z[(j * NHK + h) * NB + n];
    if (n < NNM) v = __builtin_amdgcn_exp2f(v * LOG2E);  // w = exp(z)
    wbuf[tid] = (__bf16)v;
  } else {
    int e = tid - NK * 2 * 32 * 16;
    if (e < 4 * 32 * 16) {                 // 2048 elements of w1buf
      int i  = e & 15;
      int l  = (e >> 4) & 31;
      int q  = e >> 9;
      int h  = q * 16 + (l & 15);
      int nm = ((l & 16) ? 16 : 0) + i;
      w1buf[e] = (__bf16)W1[h * NNM + nm];
    }
  }
}

// ---------------------------------------------------------------------------
// Main fused kernel: 8 waves / block, each wave owns 16 batch rows.
//   A-fragment layout used here (16-bit A, 16x32): lanes 0-15 -> row M=l, K 0..15;
//   lanes 16-31 -> row M=l-16, K 16..31 (element i: K = base + i).
//   C/D layout: VGPR r, lane l -> row M = r + (l>=16?8:0), col N = l&15.
//
// Epilogue math: with S_k = sum_h exp(be*(a+t)),
//   g_k = log(S_k)/be ; exp(-be*g_k) = 1/S_k
//   y   = -(1/be) * log( sum_k 1/S_k ) + aux      (no per-head logs needed)
// be = e^-1, |be*a| <~ 15, so unstabilized exp sums are f32-safe.
// ---------------------------------------------------------------------------
__global__ void __launch_bounds__(256)
smmnet_kernel(const float* __restrict__ x, const float* __restrict__ t,
              const float* __restrict__ beta_p, const float* __restrict__ b1,
              const float* __restrict__ W2, const float* __restrict__ b2,
              const __bf16* __restrict__ wbuf, const __bf16* __restrict__ w1buf,
              float* __restrict__ out) {
  const int lane   = threadIdx.x & 31;
  const int wave   = threadIdx.x >> 5;
  const int m0     = blockIdx.x * 128 + wave * 16;
  const int lane16 = lane & 15;
  const int kb     = (lane & 16) ? 16 : 0;  // per-half K base for A fragments

  // ---- load x row slice and build two A fragments (K 0..31, K 32..63) ----
  const float* xr = x + (size_t)(m0 + lane16) * NB;
  float xf[32];
#pragma unroll
  for (int g = 0; g < 4; ++g) {            // 4 groups of 8 contiguous floats
    const float4* p = (const float4*)(xr + (g >> 1) * 32 + kb + (g & 1) * 8);
    float4 a = p[0], bq = p[1];
    xf[g * 8 + 0] = a.x;  xf[g * 8 + 1] = a.y;
    xf[g * 8 + 2] = a.z;  xf[g * 8 + 3] = a.w;
    xf[g * 8 + 4] = bq.x; xf[g * 8 + 5] = bq.y;
    xf[g * 8 + 6] = bq.z; xf[g * 8 + 7] = bq.w;
  }
  v16bf af0, af1;
#pragma unroll
  for (int i = 0; i < 16; ++i) { af0[i] = (__bf16)xf[i]; af1[i] = (__bf16)xf[16 + i]; }

  // ---- per-lane constants (pre-scaled into exp2 domain) ----
  const float be     = __builtin_amdgcn_exp2f(beta_p[0] * LOG2E);  // e^beta > 0
  const float c1     = be * LOG2E;            // exp(be*v) = exp2(c1*v + c2)
  const float inv_bl = 1.0f / c1;
  float c2v[NK];
#pragma unroll
  for (int j = 0; j < NK; ++j) c2v[j] = t[j * NHK + lane16] * c1;
  float b1v[4], w2v[4];
#pragma unroll
  for (int q = 0; q < 4; ++q) {
    b1v[q] = b1[q * 16 + lane16];
    w2v[q] = W2[q * 16 + lane16];
  }
  const float b2v = b2[0];

  // ---- main GEMM: a[16 rows][k=8 tiles of 16 h], K=64 in 2 WMMAs/tile ----
  v8f acc[NK];
#pragma unroll
  for (int j = 0; j < NK; ++j) {
    v16bf bw0 = *(const v16bf*)(wbuf + ((size_t)(j * 2 + 0) * 32 + lane) * 16);
    v16bf bw1 = *(const v16bf*)(wbuf + ((size_t)(j * 2 + 1) * 32 + lane) * 16);
    v8f c = {};
    c = wmma_bf16(af0, bw0, c);
    c = wmma_bf16(af1, bw1, c);
    acc[j] = c;
  }

  // ---- MLP GEMM: xu = x[:,32:64] is exactly A-chunk af1; H=64 = 4 tiles ----
  v8f macc[4];
#pragma unroll
  for (int q = 0; q < 4; ++q) {
    v16bf bw = *(const v16bf*)(w1buf + ((size_t)q * 32 + lane) * 16);
    v8f c = {};
    macc[q] = wmma_bf16(af1, bw, c);
  }

  // ---- epilogue: per output row, 9 concurrent 16-lane butterfly sums ----
#pragma unroll
  for (int r = 0; r < 8; ++r) {
    // MLP partial: relu(W1 xu + b1) . W2 contribution of h = lane16 (4 tiles)
    float p = 0.0f;
#pragma unroll
    for (int q = 0; q < 4; ++q) p += fmaxf(macc[q][r] + b1v[q], 0.0f) * w2v[q];

    // per-head exp terms: s[j] = exp2(c1*a + c2) for h = lane16
    float s[NK];
#pragma unroll
    for (int j = 0; j < NK; ++j)
      s[j] = __builtin_amdgcn_exp2f(fmaf(acc[j][r], c1, c2v[j]));

    // shared butterfly: sum over the 16 h-lanes (halves stay independent)
#pragma unroll
    for (int m = 1; m < 16; m <<= 1) {
      p += __shfl_xor(p, m);
#pragma unroll
      for (int j = 0; j < NK; ++j) s[j] += __shfl_xor(s[j], m);
    }

    // smooth-min over k: y0 = -(1/be) * ln( sum_k 1/S_k )
    float hsum = 0.0f;
#pragma unroll
    for (int j = 0; j < NK; ++j) hsum += __builtin_amdgcn_rcpf(s[j]);
    const float y = -__builtin_amdgcn_logf(hsum) * inv_bl + (p + b2v);

    // row = m0 + r (half 0) / m0 + 8 + r (half 1); one writer lane per half
    if (lane == 0)       out[m0 + r]     = y;
    else if (lane == 16) out[m0 + 8 + r] = y;
  }
}

extern "C" void kernel_launch(void* const* d_in, const int* in_sizes, int n_in,
                              void* d_out, int out_size, void* d_ws, size_t ws_size,
                              hipStream_t stream) {
  const float* x    = (const float*)d_in[0];
  const float* z    = (const float*)d_in[1];
  const float* t    = (const float*)d_in[2];
  const float* beta = (const float*)d_in[3];
  const float* W1   = (const float*)d_in[4];
  const float* b1   = (const float*)d_in[5];
  const float* W2   = (const float*)d_in[6];
  const float* b2   = (const float*)d_in[7];

  __bf16* wbuf  = (__bf16*)d_ws;          // 8192 bf16 = 16 KB
  __bf16* w1buf = wbuf + 8192;            // 2048 bf16 =  4 KB

  prep_kernel<<<40, 256, 0, stream>>>(z, W1, wbuf, w1buf);

  const int rows = in_sizes[0] / NB;      // B = 1048576
  smmnet_kernel<<<rows / 128, 256, 0, stream>>>(
      x, t, beta, b1, W2, b2, wbuf, w1buf, (float*)d_out);
}
